// Attention_68882685493579
// MI455X (gfx1250) — matrix operand order repaired and checked by reading, not compiled, not run on hardware
//
#include <hip/hip_runtime.h>
#include <hip/hip_bf16.h>

typedef __attribute__((ext_vector_type(16))) _Float16 v16h;
typedef __attribute__((ext_vector_type(8)))  _Float16 v8h;
typedef __attribute__((ext_vector_type(8)))  float    v8f;

#define DIM        512
#define HEADS      8
#define DIM_HEAD   64
#define DIM_INNER  512
#define NSEQ       1024
#define NKV        2
#define JROWS      1026          // NSEQ + NKV
#define JPADROWS   1056          // padded K/V rows (multiple of 32, >= max JLD)
#define CD_ITERS   50
#define CD_EPS     0.1f
#define LOGK       2.0794415416798357f   // ln(8)
#define LN_EPS     1e-5f
#define NE         33                    // ceil(1026/32) register elements per lane
// base-2 constants: sb2 = (s + b)/eps * log2(e);  lse = ln2*(m2 + log2(sum2))
#define SBSC       14.426950408889634f   // (1/eps)*log2(e)
#define CD_C1      (CD_EPS * LOGK)       // eps*ln(k)
#define CD_C2      (CD_EPS * 0.6931471805599453f)  // eps*ln2

// ---------------------------------------------------------------------------
// f32 -> f16 convert (plain)
// ---------------------------------------------------------------------------
__global__ void f32_to_f16_kernel(const float* __restrict__ src,
                                  _Float16* __restrict__ dst, int n) {
    int i = blockIdx.x * blockDim.x + threadIdx.x;
    if (i < n) dst[i] = (_Float16)src[i];
}

// f32 (R x C row-major) -> f16 transposed (C x R row-major)
__global__ void f32_to_f16_T_kernel(const float* __restrict__ src,
                                    _Float16* __restrict__ dst, int R, int C) {
    int idx = blockIdx.x * blockDim.x + threadIdx.x;
    if (idx >= R * C) return;
    int r = idx / C, c = idx % C;                      // coalesced reads
    dst[(size_t)c * R + r] = (_Float16)src[idx];
}

// ---------------------------------------------------------------------------
// LayerNorm: one block (256 threads) per row of 512; writes f16 normalized x
// ---------------------------------------------------------------------------
__global__ __launch_bounds__(256)
void ln_kernel(const float* __restrict__ x, const float* __restrict__ g,
               const float* __restrict__ bsh, _Float16* __restrict__ xh) {
    __shared__ float red[256];
    const int row = blockIdx.x;
    const int tid = threadIdx.x;
    const float* xr = x + (size_t)row * DIM;
    float v0 = xr[tid], v1 = xr[tid + 256];

    red[tid] = v0 + v1;
    __syncthreads();
    for (int o = 128; o; o >>= 1) { if (tid < o) red[tid] += red[tid + o]; __syncthreads(); }
    float mu = red[0] * (1.0f / DIM);
    __syncthreads();

    float d0 = v0 - mu, d1 = v1 - mu;
    red[tid] = d0 * d0 + d1 * d1;
    __syncthreads();
    for (int o = 128; o; o >>= 1) { if (tid < o) red[tid] += red[tid + o]; __syncthreads(); }
    float rs = rsqrtf(red[0] * (1.0f / DIM) + LN_EPS);

    _Float16* xo = xh + (size_t)row * DIM;
    xo[tid]       = (_Float16)(d0 * rs * g[tid]       + bsh[tid]);
    xo[tid + 256] = (_Float16)(d1 * rs * g[tid + 256] + bsh[tid + 256]);
}

// ---------------------------------------------------------------------------
// WMMA GEMM, B pre-transposed: C(f32, MxN) = A(f16, MxK) * BT(f16, NxK)^T
// 256 threads = 8 wave32; each wave computes a 16x64 quad of tiles so one
// A fragment feeds 4 WMMAs. All fragment loads are contiguous b128s.
// ---------------------------------------------------------------------------
__global__ __launch_bounds__(256)
void gemm_f16t_kernel(const _Float16* __restrict__ A, const _Float16* __restrict__ BT,
                      float* __restrict__ C, int M, int N, int K) {
    const int lane = threadIdx.x & 31;
    const int wv   = threadIdx.x >> 5;
    const int nq   = N >> 6;                           // 64-col quads per row
    const int quads = (M >> 4) * nq;
    const int q = blockIdx.x * 8 + wv;
    if (q >= quads) return;
    const int ti  = q / nq;
    const int tj0 = (q % nq) << 2;                     // first 16-col tile index

    const _Float16* Ab = A + (size_t)(ti * 16 + (lane & 15)) * K + ((lane >> 4) << 4);
    const int kb2 = (lane >> 4) << 4;
    const _Float16* Bb = BT + (size_t)(tj0 * 16 + (lane & 15)) * K + kb2;

    v8f c0 = {}, c1 = {}, c2 = {}, c3 = {};
    for (int k0 = 0; k0 < K; k0 += 32) {
        __builtin_prefetch(Ab + k0 + 64, 0, 1);        // global_prefetch_b8
        v8h lo = *(const v8h*)(Ab + k0);
        v8h hi = *(const v8h*)(Ab + k0 + 8);
        v16h a;
#pragma unroll
        for (int t = 0; t < 8; ++t) { a[t] = lo[t]; a[8 + t] = hi[t]; }

        v16h b0 = *(const v16h*)(Bb + k0);
        v16h b1 = *(const v16h*)(Bb + (size_t)16 * K + k0);
        v16h b2 = *(const v16h*)(Bb + (size_t)32 * K + k0);
        v16h b3 = *(const v16h*)(Bb + (size_t)48 * K + k0);
        c0 = __builtin_amdgcn_wmma_f32_16x16x32_f16(false, a, false, b0, (short)0, c0, false, false);
        c1 = __builtin_amdgcn_wmma_f32_16x16x32_f16(false, a, false, b1, (short)0, c1, false, false);
        c2 = __builtin_amdgcn_wmma_f32_16x16x32_f16(false, a, false, b2, (short)0, c2, false, false);
        c3 = __builtin_amdgcn_wmma_f32_16x16x32_f16(false, a, false, b3, (short)0, c3, false, false);
    }
    float* Cb = C + (size_t)(ti * 16 + ((lane >> 4) << 3)) * N + tj0 * 16 + (lane & 15);
#pragma unroll
    for (int r = 0; r < 8; ++r) {
        Cb[(size_t)r * N]      = c0[r];
        Cb[(size_t)r * N + 16] = c1[r];
        Cb[(size_t)r * N + 32] = c2[r];
        Cb[(size_t)r * N + 48] = c3[r];
    }
}

// ---------------------------------------------------------------------------
// Pack heads: qkv f32 (1024x1536) + null_kv ->
//   qh [h][1024][64]  f16 (pre-scaled by d^-0.5)
//   kh [h][1056][64]  f16 (rows 0..1 null, 1026..1055 zero)
//   vT [h][64][1056]  f16 TRANSPOSED (so attn.V B-fragments are contiguous)
// ---------------------------------------------------------------------------
__global__ void pack_kernel(const float* __restrict__ qkv, const float* __restrict__ nullkv,
                            _Float16* __restrict__ qh, _Float16* __restrict__ kh,
                            _Float16* __restrict__ vT) {
    int idx = blockIdx.x * blockDim.x + threadIdx.x;   // over 8*1056*64
    if (idx >= HEADS * JPADROWS * DIM_HEAD) return;
    const int d = idx & 63;
    const int j = (idx >> 6) % JPADROWS;
    const int h = idx / (JPADROWS * DIM_HEAD);

    float kv = 0.f, vv = 0.f;
    if (j < NKV) {
        kv = nullkv[((size_t)(0 * HEADS + h) * NKV + j) * DIM_HEAD + d];
        vv = nullkv[((size_t)(1 * HEADS + h) * NKV + j) * DIM_HEAD + d];
    } else if (j < JROWS) {
        const int r = j - NKV;
        kv = qkv[(size_t)r * (3 * DIM_INNER) + DIM_INNER     + h * DIM_HEAD + d];
        vv = qkv[(size_t)r * (3 * DIM_INNER) + 2 * DIM_INNER + h * DIM_HEAD + d];
    }
    kh[((size_t)h * JPADROWS + j) * DIM_HEAD + d] = (_Float16)kv;
    vT[((size_t)h * DIM_HEAD + d) * JPADROWS + j] = (_Float16)vv;
    if (j >= NKV && j < JROWS) {
        const int r = j - NKV;
        qh[((size_t)h * NSEQ + r) * DIM_HEAD + d] =
            (_Float16)(qkv[(size_t)r * (3 * DIM_INNER) + h * DIM_HEAD + d] * 0.125f);
    }
}

// ---------------------------------------------------------------------------
// Fused attention + 50-iter coordinate descent.
// Grid (64 row-blocks, 8 heads); block 512 threads = 16 wave32.
//  Phase 1: 16 waves WMMA q(16x64).K^T -> fp32 scores in LDS.
//  Phase 2: wave w owns row w. Row (<=33 elems/lane) pulled into REGISTERS;
//           49 iterations never touch memory. Two-pass base-2 logsumexp:
//           exact max (butterfly fmax) then single exp2 per element
//           (butterfly add). b = min(s+a,0); iter-1 a closed-form.
//           Final scores written as f16 into the aliased LDS region.
//  Phase 3: waves 0..3 WMMA attn(16xJ).V(Jx64); A frags = ds b128 loads,
//           B frags contiguous from transposed V.
// ---------------------------------------------------------------------------
__global__ __launch_bounds__(512)
void attn_cd_kernel(const _Float16* __restrict__ qh, const _Float16* __restrict__ kh,
                    const _Float16* __restrict__ vT, float* __restrict__ out_heads) {
    extern __shared__ float sls[];                     // [16][JLD] fp32, then f16 alias
    const int bi   = blockIdx.x;
    const int h    = blockIdx.y;
    const int wv   = threadIdx.x >> 5;
    const int lane = threadIdx.x & 31;

    const int numJT = bi + 2;
    int JLD = numJT * 16;
    if (JLD & 31) JLD += 16;                           // multiple of 32

    // q A-fragments (16x32 each, K=0..31 / 32..63)
    const _Float16* qb = qh + ((size_t)h * NSEQ + bi * 16) * DIM_HEAD
                            + (size_t)(lane & 15) * DIM_HEAD + ((lane >> 4) << 4);
    v16h qa0, qa1;
    {
        v8h l0 = *(const v8h*)(qb);      v8h h0 = *(const v8h*)(qb + 8);
        v8h l1 = *(const v8h*)(qb + 32); v8h h1 = *(const v8h*)(qb + 40);
#pragma unroll
        for (int t = 0; t < 8; ++t) {
            qa0[t] = l0[t]; qa0[8 + t] = h0[t];
            qa1[t] = l1[t]; qa1[8 + t] = h1[t];
        }
    }

    const int c0 = lane & 15;
    const int kb = (lane >> 4) << 4;                   // B K-split (0/16)
    const int rb = (lane >> 4) << 3;                   // C row-split (0/8)

    // --- Phase 1: sim tiles -> LDS (fp32) ---
    const _Float16* kbase = kh + (size_t)h * JPADROWS * DIM_HEAD;
    for (int jt = wv; jt < numJT; jt += 16) {
        const _Float16* kp = kbase + (size_t)(jt * 16 + c0) * DIM_HEAD;
        v16h b0 = *(const v16h*)(kp + kb);
        v16h b1 = *(const v16h*)(kp + 32 + kb);
        v8f cc = {};
        cc = __builtin_amdgcn_wmma_f32_16x16x32_f16(false, qa0, false, b0, (short)0, cc, false, false);
        cc = __builtin_amdgcn_wmma_f32_16x16x32_f16(false, qa1, false, b1, (short)0, cc, false, false);
#pragma unroll
        for (int r = 0; r < 8; ++r) sls[(rb + r) * JLD + jt * 16 + c0] = cc[r];
    }
    __syncthreads();

    // --- Phase 2: coordinate descent in registers, one wave per row ---
    {
        const int Jrow = bi * 16 + wv + 3;             // cols 0..row+2 attend
        float* srow = sls + wv * JLD;
        float sreg[NE], sbreg[NE];
#pragma unroll
        for (int e = 0; e < NE; ++e) {
            int j = lane + e * 32;
            sreg[e] = (j < Jrow) ? srow[j] : -1e30f;   // sentinel -> exp2 == 0
        }
        __syncthreads();                               // fp32 region now dead

        float a = CD_C1 - CD_C2 * __log2f((float)Jrow);  // iteration 1 closed form
        for (int it = 0; it < CD_ITERS - 1; ++it) {
            float m = -3.0e37f;
#pragma unroll
            for (int e = 0; e < NE; ++e) {
                float s  = sreg[e];
                float sb = (s + fminf(s + a, 0.f)) * SBSC;
                sbreg[e] = sb;
                m = fmaxf(m, sb);
            }
#pragma unroll
            for (int o = 16; o; o >>= 1) m = fmaxf(m, __shfl_xor(m, o, 32));
            float sum = 0.f;
#pragma unroll
            for (int e = 0; e < NE; ++e) sum += exp2f(sbreg[e] - m);
#pragma unroll
            for (int o = 16; o; o >>= 1) sum += __shfl_xor(sum, o, 32);
            a = CD_C1 - CD_C2 * (m + __log2f(sum));
        }
        // final scores: exp((s+a+b)/eps), masked -> 0; write f16 into alias
        _Float16* shrow = ((_Float16*)sls) + wv * JLD;
#pragma unroll
        for (int e = 0; e < NE; ++e) {
            int j = lane + e * 32;
            if (j < JLD) {
                float v = 0.f;
                if (j < Jrow) {
                    float s = sreg[e];
                    v = exp2f((s + a + fminf(s + a, 0.f)) * SBSC);
                }
                shrow[j] = (_Float16)v;
            }
        }
    }
    __syncthreads();

    // --- Phase 3: out = attn . V ; waves 0..3 own 16-col slices of d=64 ---
    if (wv < 4) {
        const int nb = wv * 16;
        const _Float16* sh = (const _Float16*)sls;
        const _Float16* pa0 = sh + (size_t)(lane & 15) * JLD + ((lane >> 4) << 4);
        const _Float16* vb  = vT + ((size_t)h * DIM_HEAD + nb + c0) * JPADROWS + kb;
        v8f cc = {};
        for (int j0 = 0; j0 < JLD; j0 += 32) {
            v8h lo = *(const v8h*)(pa0 + j0);          // ds_load_b128
            v8h hi = *(const v8h*)(pa0 + j0 + 8);
            v16h af;
#pragma unroll
            for (int t = 0; t < 8; ++t) { af[t] = lo[t]; af[8 + t] = hi[t]; }
            v16h bf = *(const v16h*)(vb + j0);         // contiguous from V^T
            cc = __builtin_amdgcn_wmma_f32_16x16x32_f16(false, af, false, bf, (short)0, cc, false, false);
        }
        float* ob = out_heads + (size_t)(bi * 16 + rb) * DIM_INNER + h * DIM_HEAD + nb + c0;
#pragma unroll
        for (int r = 0; r < 8; ++r) ob[(size_t)r * DIM_INNER] = cc[r];
    }
}

// ---------------------------------------------------------------------------
// Launch
// ---------------------------------------------------------------------------
extern "C" void kernel_launch(void* const* d_in, const int* in_sizes, int n_in,
                              void* d_out, int out_size, void* d_ws, size_t ws_size,
                              hipStream_t stream) {
    const float* x       = (const float*)d_in[0];
    const float* w_qkv   = (const float*)d_in[1];
    const float* w_out   = (const float*)d_in[2];
    const float* null_kv = (const float*)d_in[3];
    const float* ln_g    = (const float*)d_in[4];
    const float* ln_b    = (const float*)d_in[5];
    float* out = (float*)d_out;
    char*  ws  = (char*)d_ws;

    // workspace layout (bytes, all 256-aligned)
    const size_t OFF_WQKVT = 0;                                     // 1536x512 f16 (transposed)
    const size_t OFF_WOUTT = OFF_WQKVT + (size_t)3 * DIM_INNER * DIM * 2;
    const size_t OFF_XH    = OFF_WOUTT + (size_t)DIM * DIM_INNER * 2;
    const size_t OFF_QKV   = OFF_XH    + (size_t)NSEQ * DIM * 2;
    const size_t OFF_QH    = OFF_QKV   + (size_t)NSEQ * 3 * DIM_INNER * 4;
    const size_t OFF_KH    = OFF_QH    + (size_t)HEADS * NSEQ * DIM_HEAD * 2;
    const size_t OFF_VT    = OFF_KH    + (size_t)HEADS * JPADROWS * DIM_HEAD * 2;
    const size_t OFF_OH    = OFF_VT    + (size_t)HEADS * DIM_HEAD * JPADROWS * 2;
    const size_t OFF_OHH   = OFF_OH    + (size_t)NSEQ * DIM_INNER * 4;

    _Float16* wqkvT = (_Float16*)(ws + OFF_WQKVT);
    _Float16* woutT = (_Float16*)(ws + OFF_WOUTT);
    _Float16* xh    = (_Float16*)(ws + OFF_XH);
    float*    qkv   = (float*)   (ws + OFF_QKV);
    _Float16* qh    = (_Float16*)(ws + OFF_QH);
    _Float16* kh    = (_Float16*)(ws + OFF_KH);
    _Float16* vT    = (_Float16*)(ws + OFF_VT);
    float*    oh    = (float*)   (ws + OFF_OH);
    _Float16* ohh   = (_Float16*)(ws + OFF_OHH);

    const int n_wqkv = DIM * 3 * DIM_INNER;
    const int n_wout = DIM_INNER * DIM;
    f32_to_f16_T_kernel<<<(n_wqkv + 255) / 256, 256, 0, stream>>>(w_qkv, wqkvT, DIM, 3 * DIM_INNER);
    f32_to_f16_T_kernel<<<(n_wout + 255) / 256, 256, 0, stream>>>(w_out, woutT, DIM_INNER, DIM);

    ln_kernel<<<NSEQ, 256, 0, stream>>>(x, ln_g, ln_b, xh);

    // QKV projection: (1024x512) * (512x1536)
    {
        int quads = (NSEQ / 16) * (3 * DIM_INNER / 64);
        gemm_f16t_kernel<<<(quads + 7) / 8, 256, 0, stream>>>(xh, wqkvT, qkv,
                                                              NSEQ, 3 * DIM_INNER, DIM);
    }

    {
        int n = HEADS * JPADROWS * DIM_HEAD;
        pack_kernel<<<(n + 255) / 256, 256, 0, stream>>>(qkv, null_kv, qh, kh, vT);
    }

    // fused attention + CD; dynamic LDS = 16 rows * 1056 cols * 4B (<= 320KB WGP)
    attn_cd_kernel<<<dim3(NSEQ / 16, HEADS), 512, 16 * 1056 * sizeof(float), stream>>>(
        qh, kh, vT, oh);

    // output projection: (1024x512) * (512x512)
    {
        int n = NSEQ * DIM_INNER;
        f32_to_f16_kernel<<<(n + 255) / 256, 256, 0, stream>>>(oh, ohh, n);
        int quads = (NSEQ / 16) * (DIM / 64);
        gemm_f16t_kernel<<<(quads + 7) / 8, 256, 0, stream>>>(ohh, woutT, out,
                                                              NSEQ, DIM, DIM_INNER);
    }
}
